// TripletLoss_35966056137080
// MI455X (gfx1250) — compile-verified, matrix operand order checked by reading, not run on hardware
//
#include <hip/hip_runtime.h>

typedef __attribute__((ext_vector_type(16))) _Float16 v16h;
typedef __attribute__((ext_vector_type(8)))  float    v8f;

#define N_EMB   8192
#define D_EMB   512
#define KBLK    16        // D_EMB / 32 k-chunks per 16x16 tile
#define ROWH    520       // padded LDS row stride in halves (1040 B = 4 banks mod 64)
#define NEG_BIGF 1e9f
#define MARGIN_F 1.0f
#define COS_EPSF 1e-8f
#define PD_EPSF  1e-6f

// 16-byte-aligned view of a 32-byte fragment (two ds_load_b128)
struct __attribute__((packed, aligned(16))) Frag16 { v16h v; };

// generic -> LDS-relative 32-bit offset (strips the LDS aperture base)
__device__ __forceinline__ unsigned lds_off(const void* p) {
  return (unsigned)(uintptr_t)(__attribute__((address_space(3))) const void*)p;
}

// Stage 128 bytes of a B tile: 8 async 16B transfers. The immediate offset is
// added to BOTH the LDS and the global address by hardware, and both sides
// have stride 128 in 's', so one base pair covers all 8.
__device__ __forceinline__ void stage_tile_async(unsigned ldsDst,
                                                 const _Float16* gBase,
                                                 unsigned gOff) {
  unsigned long long sb = (unsigned long long)(uintptr_t)gBase;
#define ASYNC16(IMM)                                                          \
  asm volatile("global_load_async_to_lds_b128 %0, %1, %2 offset:" #IMM       \
               :: "v"(ldsDst), "v"(gOff), "s"(sb) : "memory")
  ASYNC16(0);   ASYNC16(128); ASYNC16(256); ASYNC16(384);
  ASYNC16(512); ASYNC16(640); ASYNC16(768); ASYNC16(896);
#undef ASYNC16
}

// ---------------------------------------------------------------------------
// Kernel 1: fp32 row norms -> f16 unit vectors (one wave32 per row)
// ---------------------------------------------------------------------------
__global__ __launch_bounds__(256) void tl_normalize_kernel(
    const float* __restrict__ emb, _Float16* __restrict__ unitH) {
  const int wave = threadIdx.x >> 5;
  const int ln   = threadIdx.x & 31;
  const int row  = blockIdx.x * 8 + wave;
  const float* r = emb + (size_t)row * D_EMB;

  float x[16];
  float s = 0.0f;
#pragma unroll
  for (int t = 0; t < 16; ++t) {
    x[t] = r[ln + 32 * t];
    s += x[t] * x[t];
  }
#pragma unroll
  for (int m = 16; m >= 1; m >>= 1) s += __shfl_xor(s, m, 32);

  const float inv = 1.0f / fmaxf(sqrtf(s), COS_EPSF);
  _Float16* o = unitH + (size_t)row * D_EMB;
#pragma unroll
  for (int t = 0; t < 16; ++t) o[ln + 32 * t] = (_Float16)(x[t] * inv);
}

// ---------------------------------------------------------------------------
// Kernel 2: fused dist-matrix + batch-hard mining.
// Block = 4 waves = 64 i-rows. B j-tiles (16 rows x 512 halves = 16KB) are
// staged once per block via async-to-LDS, double-buffered; each wave runs
// 16x wmma_f32_16x16x32_f16 per tile out of LDS. sched_group_barrier forces
// a depth-4 DS-read/WMMA pipeline so waits become partial instead of drains.
// D layout: acc[r] at lane ln is D(m = r + 8*(ln>>4), n = ln&15).
// ---------------------------------------------------------------------------
__global__ __launch_bounds__(128)
__attribute__((amdgpu_waves_per_eu(1)))
void tl_mine_kernel(
    const _Float16* __restrict__ unitH, const int* __restrict__ labels,
    int* __restrict__ posIdx, int* __restrict__ negIdx, int* __restrict__ validF) {
  __shared__ _Float16 sb[2][16 * ROWH];   // 2 x 16.25 KB, double buffered

  const int t    = threadIdx.x;
  const int wave = t >> 5;
  const int ln   = t & 31;
  const int i0   = (blockIdx.x * 4 + wave) * 16;

  const int nsub = ln & 15;   // row-in-tile for operand loads; N column in D
  const int hi   = ln >> 4;   // K-half for operand loads; row group (M += 8) in D

  // Preload A fragments for this i-tile (register-resident across the j loop).
  v16h afrag[KBLK];
  const _Float16* arow = unitH + (size_t)(i0 + nsub) * D_EMB + hi * 16;
#pragma unroll
  for (int kb = 0; kb < KBLK; ++kb)
    afrag[kb] = *(const v16h*)(arow + kb * 32);

  // Labels of my 8 output rows.
  int labI[8];
#pragma unroll
  for (int r = 0; r < 8; ++r) labI[r] = labels[i0 + r + 8 * hi];

  float bestP[8], bestN[8];
  int   bpi[8],  bni[8];
#pragma unroll
  for (int r = 0; r < 8; ++r) {
    bestP[r] = -NEG_BIGF; bestN[r] = NEG_BIGF; bpi[r] = 0; bni[r] = 0;
  }

  // Staging map: thread t copies row sr = t>>3, units c = (t&7) + 8s (16B each).
  const int sr = t >> 3;
  const int sc = t & 7;
  const unsigned gOff    = (unsigned)(sr * (D_EMB * 2) + sc * 16);
  const unsigned ldsRowA = lds_off(&sb[0][0]) + (unsigned)(sr * (ROWH * 2) + sc * 16);
  const unsigned ldsRowB = lds_off(&sb[1][0]) + (unsigned)(sr * (ROWH * 2) + sc * 16);

  // Prologue: stage tile 0 into buffer 0.
  stage_tile_async(ldsRowA, unitH, gOff);

  for (int j0 = 0, it = 0; j0 < N_EMB; j0 += 16, ++it) {
    // My async writes into buf[it&1] done:
    asm volatile("s_wait_asynccnt 0x0" ::: "memory");
    // All waves' writes visible; all waves done reading buf[(it+1)&1] (iter it-1):
    __syncthreads();
    // Prefetch next tile into the other buffer while we compute this one.
    if (j0 + 16 < N_EMB)
      stage_tile_async((it & 1) ? ldsRowA : ldsRowB,
                       unitH + (size_t)(j0 + 16) * D_EMB, gOff);

    const int labJ = labels[j0 + nsub];
    const _Float16* bufc = &sb[it & 1][nsub * ROWH + hi * 16];

#define BF(KB) (((const Frag16*)(bufc + (KB) * 32))->v)
#define WM(KB, BV)                                                            \
  acc = __builtin_amdgcn_wmma_f32_16x16x32_f16(false, afrag[KB], false, (BV), \
                                               (short)0, acc, false, false)
    v8f acc = {};
    // Depth-4 software pipeline on B fragments (rotated registers b0..b3).
    v16h b0 = BF(0), b1 = BF(1), b2 = BF(2), b3 = BF(3);
    WM(0,  b0); b0 = BF(4);
    WM(1,  b1); b1 = BF(5);
    WM(2,  b2); b2 = BF(6);
    WM(3,  b3); b3 = BF(7);
    WM(4,  b0); b0 = BF(8);
    WM(5,  b1); b1 = BF(9);
    WM(6,  b2); b2 = BF(10);
    WM(7,  b3); b3 = BF(11);
    WM(8,  b0); b0 = BF(12);
    WM(9,  b1); b1 = BF(13);
    WM(10, b2); b2 = BF(14);
    WM(11, b3); b3 = BF(15);
    WM(12, b0);
    WM(13, b1);
    WM(14, b2);
    WM(15, b3);
#undef BF
#undef WM

#if __has_builtin(__builtin_amdgcn_sched_group_barrier)
    // Issue pattern for this scheduling region:
    //   8 DS-reads (4 fragments) up front, then {1 WMMA, 2 DS-reads} x12,
    //   then the last 4 WMMAs. Keeps 3-4 ds_loads in flight behind each WMMA,
    //   turning s_wait_dscnt 0x0 drains into partial waits.
#define SGB __builtin_amdgcn_sched_group_barrier
    SGB(0x100, 8, 0);                       // 8 DS reads
    SGB(0x008, 1, 0); SGB(0x100, 2, 0);     // wmma 0
    SGB(0x008, 1, 0); SGB(0x100, 2, 0);     // wmma 1
    SGB(0x008, 1, 0); SGB(0x100, 2, 0);     // wmma 2
    SGB(0x008, 1, 0); SGB(0x100, 2, 0);     // wmma 3
    SGB(0x008, 1, 0); SGB(0x100, 2, 0);     // wmma 4
    SGB(0x008, 1, 0); SGB(0x100, 2, 0);     // wmma 5
    SGB(0x008, 1, 0); SGB(0x100, 2, 0);     // wmma 6
    SGB(0x008, 1, 0); SGB(0x100, 2, 0);     // wmma 7
    SGB(0x008, 1, 0); SGB(0x100, 2, 0);     // wmma 8
    SGB(0x008, 1, 0); SGB(0x100, 2, 0);     // wmma 9
    SGB(0x008, 1, 0); SGB(0x100, 2, 0);     // wmma 10
    SGB(0x008, 1, 0); SGB(0x100, 2, 0);     // wmma 11
    SGB(0x008, 4, 0);                       // wmma 12..15
#undef SGB
#endif

#pragma unroll
    for (int r = 0; r < 8; ++r) {
      const float dist = 1.0f - acc[r];
      const int gi = i0 + r + 8 * hi;
      const int gj = j0 + nsub;
      const bool same = (labI[r] == labJ);
      const bool self = (gi == gj);
      const float pv = (same && !self) ? dist : -NEG_BIGF;
      if (pv > bestP[r]) { bestP[r] = pv; bpi[r] = gj; }   // strict > : first idx on ties
      const float nv = (!same) ? dist : NEG_BIGF;
      if (nv < bestN[r]) { bestN[r] = nv; bni[r] = gj; }
    }
  }

  // Argmax/argmin reduce across the 16 lanes holding the same row
  // (xor masks 8..1 stay within each 16-lane group). Ties -> smaller index.
#pragma unroll
  for (int r = 0; r < 8; ++r) {
#pragma unroll
    for (int m = 8; m >= 1; m >>= 1) {
      float ovP = __shfl_xor(bestP[r], m, 32);
      int   oiP = __shfl_xor(bpi[r],   m, 32);
      if (ovP > bestP[r] || (ovP == bestP[r] && oiP < bpi[r])) { bestP[r] = ovP; bpi[r] = oiP; }
      float ovN = __shfl_xor(bestN[r], m, 32);
      int   oiN = __shfl_xor(bni[r],   m, 32);
      if (ovN < bestN[r] || (ovN == bestN[r] && oiN < bni[r])) { bestN[r] = ovN; bni[r] = oiN; }
    }
  }

  if (nsub == 0) {
#pragma unroll
    for (int r = 0; r < 8; ++r) {
      const int gi = i0 + r + 8 * hi;
      posIdx[gi] = bpi[r];
      negIdx[gi] = bni[r];
      validF[gi] = (bestP[r] > -0.5f * NEG_BIGF) && (bestN[r] < 0.5f * NEG_BIGF) ? 1 : 0;
    }
  }
}

// ---------------------------------------------------------------------------
// Kernel 3: exact fp32 triplet margin per row (one wave32 per row)
// ---------------------------------------------------------------------------
__global__ __launch_bounds__(256) void tl_triplet_kernel(
    const float* __restrict__ emb, const int* __restrict__ posIdx,
    const int* __restrict__ negIdx, const int* __restrict__ validF,
    float* __restrict__ perRow, float* __restrict__ cntRow) {
  const int wave = threadIdx.x >> 5;
  const int ln   = threadIdx.x & 31;
  const int row  = blockIdx.x * 8 + wave;

  const float* a = emb + (size_t)row * D_EMB;
  const float* p = emb + (size_t)posIdx[row] * D_EMB;
  const float* g = emb + (size_t)negIdx[row] * D_EMB;

  float sap = 0.0f, san = 0.0f;
#pragma unroll
  for (int t = 0; t < 16; ++t) {
    const int k = ln + 32 * t;
    const float av = a[k];
    const float dp = av - p[k] + PD_EPSF;
    const float dn = av - g[k] + PD_EPSF;
    sap += dp * dp;
    san += dn * dn;
  }
#pragma unroll
  for (int m = 16; m >= 1; m >>= 1) {
    sap += __shfl_xor(sap, m, 32);
    san += __shfl_xor(san, m, 32);
  }
  if (ln == 0) {
    const float per = fmaxf(sqrtf(sap) - sqrtf(san) + MARGIN_F, 0.0f);
    const float v = validF[row] ? 1.0f : 0.0f;
    perRow[row] = per * v;
    cntRow[row] = v;
  }
}

// ---------------------------------------------------------------------------
// Kernel 4: deterministic fixed-order reduction -> mean
// ---------------------------------------------------------------------------
__global__ __launch_bounds__(256) void tl_reduce_kernel(
    const float* __restrict__ perRow, const float* __restrict__ cntRow,
    float* __restrict__ out) {
  __shared__ float ss[256];
  __shared__ float sc[256];
  const int t = threadIdx.x;
  float s = 0.0f, c = 0.0f;
  for (int k = t; k < N_EMB; k += 256) { s += perRow[k]; c += cntRow[k]; }
  ss[t] = s; sc[t] = c;
  __syncthreads();
  for (int m = 128; m >= 1; m >>= 1) {
    if (t < m) { ss[t] += ss[t + m]; sc[t] += sc[t + m]; }
    __syncthreads();
  }
  if (t == 0) out[0] = ss[0] / fmaxf(sc[0], 1.0f);
}

// ---------------------------------------------------------------------------
extern "C" void kernel_launch(void* const* d_in, const int* in_sizes, int n_in,
                              void* d_out, int out_size, void* d_ws, size_t ws_size,
                              hipStream_t stream) {
  const float* emb    = (const float*)d_in[0];   // [8192, 512] fp32
  const int*   labels = (const int*)d_in[1];     // [8192] int32
  float*       out    = (float*)d_out;           // scalar fp32

  // Workspace layout (~8.55 MB total)
  uint8_t* ws = (uint8_t*)d_ws;
  size_t off = 0;
  _Float16* unitH = (_Float16*)(ws + off); off += (size_t)N_EMB * D_EMB * sizeof(_Float16);
  int* posIdx  = (int*)(ws + off);   off += (size_t)N_EMB * sizeof(int);
  int* negIdx  = (int*)(ws + off);   off += (size_t)N_EMB * sizeof(int);
  int* validF  = (int*)(ws + off);   off += (size_t)N_EMB * sizeof(int);
  float* perRow = (float*)(ws + off); off += (size_t)N_EMB * sizeof(float);
  float* cntRow = (float*)(ws + off); off += (size_t)N_EMB * sizeof(float);
  (void)ws_size; (void)in_sizes; (void)n_in; (void)out_size;

  tl_normalize_kernel<<<N_EMB / 8, 256, 0, stream>>>(emb, unitH);
  tl_mine_kernel<<<(N_EMB / 16) / 4, 128, 0, stream>>>(unitH, labels, posIdx, negIdx, validF);
  tl_triplet_kernel<<<N_EMB / 8, 256, 0, stream>>>(emb, posIdx, negIdx, validF, perRow, cntRow);
  tl_reduce_kernel<<<1, 256, 0, stream>>>(perRow, cntRow, out);
}
